// GatedLinearAttention_16896401342973
// MI455X (gfx1250) — compile-verified, matrix operand order checked by reading, not run on hardware
//
#include <hip/hip_runtime.h>

// Gated linear attention for MI455X (gfx1250), wave32 + WMMA bf16 + TDM.
//
//  k0_convert : x -> bf16, Wq/Wk/Wv -> bf16 transposed (n-major) for B-fragments
//  k1_qkv     : Q/K/V = x@W + b. Block tile 128x64, A/B k-slices staged in LDS
//               by the Tensor Data Mover (tensor_load_to_lds, double buffered,
//               s_wait_tensorcnt), WMMA bf16 from LDS.
//  k2_gate    : alpha = 0.9*sigmoid(silu(x@Wg1+bg1)@Wg2+bg2)+0.1  (fp32 VALU)
//  k3_chunk   : per (b,h,chunk): decay cumprods, softmax scores (WMMA),
//               within-chunk gated-linear scores (WMMA), combined P@V (WMMA),
//               chunk summary U = diag(total) * (k/B)^T @ V (WMMA)
//  k4_scan    : per (b,h): sequential chunk scan  S <- total (.) S + U,
//               inter = (q (.) B) @ S  (WMMA) accumulated into out.

#define DEVFN __device__ __forceinline__

typedef unsigned short u16;
typedef __attribute__((ext_vector_type(16))) __bf16 bf16x16;
typedef __attribute__((ext_vector_type(8)))  float  f32x8;
typedef __attribute__((ext_vector_type(8)))  unsigned short ushort8;
typedef __attribute__((ext_vector_type(4)))  unsigned int u32x4;
typedef __attribute__((ext_vector_type(8)))  int i32x8;
typedef __attribute__((ext_vector_type(4)))  int i32x4;

union Frag { bf16x16 v; u16 u[16]; ushort8 h[2]; };

constexpr int B_ = 2, L_ = 2048, DIM_ = 1024, H_ = 16, D_ = 64, C_ = 64, NC_ = 32;
constexpr int TOK    = B_ * L_;        // 4096 tokens
constexpr int NCHUNK = B_ * H_ * NC_;  // 1024 (b,h,chunk) tiles

DEVFN float bf2f(u16 h) { unsigned x = ((unsigned)h) << 16; float f; __builtin_memcpy(&f, &x, 4); return f; }
DEVFN u16  f2bf(float f) { unsigned x; __builtin_memcpy(&x, &f, 4); x += 0x7FFFu + ((x >> 16) & 1u); return (u16)(x >> 16); }

// K index held by element i of a 16x16x32 bf16 A/B fragment (ISA 7.12.2):
// lanes 0-15: K in {0..7, 16..23}; lanes 16-31: K in {8..15, 24..31}.
DEVFN int kidx(int i, int half) { return ((i < 8) ? 0 : 16) + (half << 3) + (i & 7); }

// A fragment, element (m,k) at p[(m0+m)*stride + k0 + k]  (row-major MxK, contiguous K)
DEVFN bf16x16 loadA_rm(const u16* p, int stride, int m0, int k0, int mr, int half) {
  Frag f;
  const u16* q = p + (m0 + mr) * stride + k0 + half * 8;  // 16B aligned by construction
  f.h[0] = *(const ushort8*)(q);
  f.h[1] = *(const ushort8*)(q + 16);
  return f.v;
}
// B fragment, element (k,n) at p[(n0+n)*stride + k0 + k]  (n-major storage, contiguous K)
DEVFN bf16x16 loadB_cm(const u16* p, int stride, int k0, int n0, int mr, int half) {
  Frag f;
  const u16* q = p + (n0 + mr) * stride + k0 + half * 8;
  f.h[0] = *(const ushort8*)(q);
  f.h[1] = *(const ushort8*)(q + 16);
  return f.v;
}

DEVFN f32x8 wmma_bf16(bf16x16 a, bf16x16 b, f32x8 c) {
  return __builtin_amdgcn_wmma_f32_16x16x32_bf16(false, a, false, b, (short)0, c, false, false);
}

// ---- Tensor Data Mover: 2D tile (rows x cols bf16, row stride in elements)
// D# packing per CDNA5 ISA 8.3/8.4. amdgpu-toolchain 6-arg builtin form.
DEVFN void tdm_load_2d(unsigned lds_off, const u16* gptr, int rows, int cols, int stride_elems) {
  unsigned long long ga = (unsigned long long)gptr;
  u32x4 g0 = {};
  g0[0] = 1u;                                             // count=1 (valid), user mode
  g0[1] = lds_off;                                        // lds_addr (bytes)
  g0[2] = (unsigned)ga;                                   // global_addr[31:0]
  g0[3] = (unsigned)((ga >> 32) & 0x01FFFFFFu) | (2u << 30);  // addr[56:32], type=2
  i32x8 g1 = {};
  g1[0] = 1 << 16;                                        // data_size=1 -> 2 bytes
  g1[1] = (int)(((unsigned)stride_elems & 0xFFFFu) << 16);      // tensor_dim0[15:0]
  g1[2] = (int)((((unsigned)stride_elems >> 16) & 0xFFFFu) | (0xFFFFu << 16)); // dim0 hi | tensor_dim1 lo
  g1[3] = (int)(((unsigned)cols & 0xFFFFu) << 16);        // tensor_dim1 hi=0 | tile_dim0
  g1[4] = (int)((unsigned)rows & 0xFFFFu);                // tile_dim1 | tile_dim2=0
  g1[5] = (int)stride_elems;                              // tensor_dim0_stride[31:0]
  g1[6] = 0;
  g1[7] = 0;
  i32x4 g2 = {};
  g2[0] = 1;                                              // tensor_dim2 = 1 (unused)
  i32x4 g3 = {};
  i32x8 g4 = {};                                          // extra group (zero-filled)
  __builtin_amdgcn_tensor_load_to_lds(g0, g1, g2, g3, g4, 0);
}

DEVFN unsigned lds_off32(const void* p) { return (unsigned)(unsigned long long)p; }

// ---------------------------------------------------------------- k0: convert
__global__ void k0_convert(const float* __restrict__ x, const float* __restrict__ Wq,
                           const float* __restrict__ Wk, const float* __restrict__ Wv,
                           u16* __restrict__ xb, u16* __restrict__ wt) {
  long i = (long)blockIdx.x * blockDim.x + threadIdx.x;
  const long NX = (long)TOK * DIM_;   // 4 Mi
  const long NW = (long)DIM_ * DIM_;  // 1 Mi
  if (i < NX) { xb[i] = f2bf(x[i]); return; }
  long r = i - NX;
  if (r >= 3 * NW) return;
  int proj = (int)(r / NW);
  long w = r - (long)proj * NW;
  int n = (int)(w >> 10), k = (int)(w & 1023);
  const float* W = (proj == 0) ? Wq : (proj == 1) ? Wk : Wv;
  wt[(long)proj * NW + (long)n * DIM_ + k] = f2bf(W[(long)k * DIM_ + n]);  // Wt[n][k] = W[k][n]
}

// --------------------------------------------- k1: QKV GEMM (TDM-staged, LDS)
__global__ void k1_qkv(const u16* __restrict__ xb, const u16* __restrict__ wt,
                       const float* __restrict__ bq, const float* __restrict__ bk,
                       const float* __restrict__ bv,
                       u16* __restrict__ qb, u16* __restrict__ kb, u16* __restrict__ vb) {
  __shared__ u16 sA[2][128 * 32];  // 8 KB x2, A k-slice (rows x 32)
  __shared__ u16 sB[2][64 * 32];   // 4 KB x2, B k-slice (n-major, 64 x 32)
  int tid = threadIdx.x;
  int lane = tid & 31, mr = lane & 15, half = lane >> 4, w = tid >> 5;
  int bid = blockIdx.x;            // 1536 blocks: 3 proj x 32 row-tiles x 16 col-tiles
  int proj = bid / 512;
  int rem  = bid % 512;
  int row0 = (rem >> 4) * 128;
  int c0   = (rem & 15) * 64;
  const u16* ap = xb + (long)row0 * DIM_;
  const u16* wp = wt + (long)proj * DIM_ * DIM_ + (long)c0 * DIM_;
  f32x8 acc[4] = {};
  if (tid < 32) {                  // wave 0 drives the TDM (EXEC ignored by TDM)
    tdm_load_2d(lds_off32(&sA[0][0]), ap +  0, 128, 32, DIM_);
    tdm_load_2d(lds_off32(&sB[0][0]), wp +  0,  64, 32, DIM_);
    tdm_load_2d(lds_off32(&sA[1][0]), ap + 32, 128, 32, DIM_);
    tdm_load_2d(lds_off32(&sB[1][0]), wp + 32,  64, 32, DIM_);
  }
  for (int i = 0; i < 32; ++i) {
    int buf = i & 1;
    if (tid < 32) {                // in-order completion: oldest fill (buf) done
      if (i < 31) __builtin_amdgcn_s_wait_tensorcnt(2);
      else        __builtin_amdgcn_s_wait_tensorcnt(0);
    }
    __syncthreads();               // LDS tiles visible to all 8 waves
    bf16x16 a = loadA_rm(&sA[buf][0], 32, w * 16, 0, mr, half);
#pragma unroll
    for (int j = 0; j < 4; ++j)
      acc[j] = wmma_bf16(a, loadB_cm(&sB[buf][0], 32, 0, j * 16, mr, half), acc[j]);
    __syncthreads();               // everyone done reading before refill
    if (tid < 32 && i + 2 < 32) {
      int kk = (i + 2) * 32;
      tdm_load_2d(lds_off32(&sA[buf][0]), ap + kk, 128, 32, DIM_);
      tdm_load_2d(lds_off32(&sB[buf][0]), wp + kk,  64, 32, DIM_);
    }
  }
  const float* bias = (proj == 0) ? bq : (proj == 1) ? bk : bv;
  u16* dst = (proj == 0) ? qb : (proj == 1) ? kb : vb;
#pragma unroll
  for (int j = 0; j < 4; ++j)
#pragma unroll
    for (int r = 0; r < 8; ++r) {
      int n = c0 + j * 16 + mr;
      int row = row0 + w * 16 + r + half * 8;
      dst[(long)row * DIM_ + n] = f2bf(acc[j][r] + bias[n]);
    }
}

// ---------------------------------------------------------------- k2: gate MLP
__global__ void k2_gate(const float* __restrict__ x, const float* __restrict__ Wg1,
                        const float* __restrict__ bg1, const float* __restrict__ Wg2,
                        const float* __restrict__ bg2, float* __restrict__ alphaw) {
  __shared__ float red[256];
  __shared__ float hl[16];
  int row = blockIdx.x;
  int tid = threadIdx.x;
  int gi = tid >> 4, tt = tid & 15;
  const float* xr = x + (long)row * DIM_;
  float p = 0.f;
  for (int k = tt; k < DIM_; k += 16) p += xr[k] * Wg1[k * 16 + gi];
  red[tid] = p;
  __syncthreads();
  if (tid < 16) {
    float s = bg1[tid];
#pragma unroll
    for (int t = 0; t < 16; ++t) s += red[tid * 16 + t];
    hl[tid] = s / (1.f + __expf(-s));  // silu
  }
  __syncthreads();
#pragma unroll
  for (int jj = 0; jj < 4; ++jj) {
    int j = tid + jj * 256;
    float g = bg2[j];
#pragma unroll
    for (int i2 = 0; i2 < 16; ++i2) g += hl[i2] * Wg2[i2 * DIM_ + j];
    alphaw[(long)row * DIM_ + j] = 0.9f / (1.f + __expf(-g)) + 0.1f;
  }
}

// ------------------------------------------------------- k3: per-chunk kernel
__global__ void k3_chunk(const u16* __restrict__ qb, const u16* __restrict__ kb,
                         const u16* __restrict__ vb, const float* __restrict__ alphaw,
                         u16* __restrict__ qtw, float* __restrict__ Uw,
                         float* __restrict__ totw, float* __restrict__ out) {
  __shared__ u16 sQ[C_ * D_];      // q -> q~ -> combined P
  __shared__ u16 sK[C_ * D_];      // k -> k/B_{s+1}
  __shared__ u16 sV[C_ * D_];
  __shared__ float sS[C_ * C_];    // alpha -> softmax scores -> probs
  __shared__ float sS2[C_ * C_];   // lin scores; later aliased as sVt/sKt (u16)
  __shared__ float sB[(C_ + 1) * D_];  // exclusive cumprod of alpha (sB[64*64+d] = total)
  int bid = blockIdx.x;
  int n = bid & 31, h = (bid >> 5) & 15, b = bid >> 9;
  int tid = threadIdx.x;
  int lane = tid & 31, mr = lane & 15, half = lane >> 4, w = tid >> 5;
  int t0 = w * 16;
  long base = ((long)b * L_ + n * C_) * DIM_ + h * D_;

  for (int idx = tid; idx < C_ * D_; idx += 128) {
    int t = idx >> 6, d = idx & 63;
    long g = base + (long)t * DIM_ + d;
    sQ[idx] = qb[g]; sK[idx] = kb[g]; sV[idx] = vb[g];
    sS[idx] = alphaw[g];             // stash alpha in sS
  }
  __syncthreads();
  if (tid < D_) {                    // exclusive decay cumprod per feature d
    float c = 1.f;
    sB[tid] = 1.f;
    for (int t = 0; t < C_; ++t) { c *= sS[t * D_ + tid]; sB[(t + 1) * D_ + tid] = c; }
  }
  __syncthreads();
  {                                  // softmax scores = q k^T / sqrt(D)
    f32x8 acc[4] = {};
#pragma unroll
    for (int kk = 0; kk < D_; kk += 32) {
      bf16x16 a = loadA_rm(sQ, D_, t0, kk, mr, half);
#pragma unroll
      for (int j = 0; j < 4; ++j)
        acc[j] = wmma_bf16(a, loadB_cm(sK, D_, kk, j * 16, mr, half), acc[j]);
    }
#pragma unroll
    for (int j = 0; j < 4; ++j)
#pragma unroll
      for (int r = 0; r < 8; ++r)
        sS[(t0 + r + half * 8) * C_ + j * 16 + mr] = acc[j][r] * 0.125f;
  }
  __syncthreads();
  for (int idx = tid; idx < C_ * D_; idx += 128) {  // in-place gating transforms
    int t = idx >> 6, d = idx & 63;
    u16 qh = f2bf(bf2f(sQ[idx]) * sB[t * D_ + d]);  // q~ = q * B_t
    sQ[idx] = qh;
    qtw[base + (long)t * DIM_ + d] = qh;            // needed again by k4
    sK[idx] = f2bf(bf2f(sK[idx]) / sB[(t + 1) * D_ + d]);  // k~ = k / B_{s+1}
  }
  __syncthreads();
  {                                  // linear scores = q~ k~^T (decay ratio <= 1, s<t)
    f32x8 acc[4] = {};
#pragma unroll
    for (int kk = 0; kk < D_; kk += 32) {
      bf16x16 a = loadA_rm(sQ, D_, t0, kk, mr, half);
#pragma unroll
      for (int j = 0; j < 4; ++j)
        acc[j] = wmma_bf16(a, loadB_cm(sK, D_, kk, j * 16, mr, half), acc[j]);
    }
#pragma unroll
    for (int j = 0; j < 4; ++j)
#pragma unroll
      for (int r = 0; r < 8; ++r)
        sS2[(t0 + r + half * 8) * C_ + j * 16 + mr] = acc[j][r];
  }
  __syncthreads();
  if (tid < C_) {                    // row softmax + combine (s<=t soft, s<t linear)
    int t = tid;
    float mx = -3.4e38f;
    for (int s = 0; s <= t; ++s) mx = fmaxf(mx, sS[t * C_ + s]);
    float sum = 0.f;
    for (int s = 0; s <= t; ++s) { float e = __expf(sS[t * C_ + s] - mx); sS[t * C_ + s] = e; sum += e; }
    float inv = 1.f / sum;
    for (int s = 0; s < C_; ++s) {
      float p = 0.f;
      if (s <= t) p = sS[t * C_ + s] * inv;
      if (s <  t) p += sS2[t * C_ + s];
      sQ[t * C_ + s] = f2bf(p);      // combined P in bf16
    }
  }
  __syncthreads();
  // transposed copies so V / k~ act as contiguous-K fragments (alias dead sS2)
  u16* sVt = (u16*)sS2;              // sVt[e*64+s] = V[s][e]
  u16* sKt = ((u16*)sS2) + C_ * D_;  // sKt[d*64+s] = k~[s][d]
  for (int idx = tid; idx < C_ * D_; idx += 128) {
    int s = idx >> 6, d = idx & 63;
    sVt[d * C_ + s] = sV[idx];
    sKt[d * C_ + s] = sK[idx];
  }
  __syncthreads();
  {                                  // out(chunk) = P @ V
    f32x8 acc[4] = {};
#pragma unroll
    for (int kk = 0; kk < C_; kk += 32) {
      bf16x16 a = loadA_rm(sQ, C_, t0, kk, mr, half);
#pragma unroll
      for (int j = 0; j < 4; ++j)
        acc[j] = wmma_bf16(a, loadB_cm(sVt, C_, kk, j * 16, mr, half), acc[j]);
    }
#pragma unroll
    for (int j = 0; j < 4; ++j)
#pragma unroll
      for (int r = 0; r < 8; ++r)
        out[base + (long)(t0 + r + half * 8) * DIM_ + j * 16 + mr] = acc[j][r];
  }
  {                                  // U = diag(total) * k~^T @ V
    f32x8 acc[4] = {};
#pragma unroll
    for (int kk = 0; kk < C_; kk += 32) {
      bf16x16 a = loadA_rm(sKt, C_, t0, kk, mr, half);
#pragma unroll
      for (int j = 0; j < 4; ++j)
        acc[j] = wmma_bf16(a, loadB_cm(sVt, C_, kk, j * 16, mr, half), acc[j]);
    }
#pragma unroll
    for (int j = 0; j < 4; ++j)
#pragma unroll
      for (int r = 0; r < 8; ++r) {
        int m = t0 + r + half * 8;
        Uw[(long)bid * (D_ * D_) + m * D_ + j * 16 + mr] = sB[C_ * D_ + m] * acc[j][r];
      }
  }
  if (tid < D_) totw[bid * D_ + tid] = sB[C_ * D_ + tid];
}

// --------------------------------------------------- k4: inter-chunk scan of S
__global__ void k4_scan(const u16* __restrict__ qtw, const float* __restrict__ Uw,
                        const float* __restrict__ totw, float* __restrict__ out) {
  __shared__ float S[D_ * D_];
  __shared__ u16   SbT[D_ * D_];     // SbT[e*64+d] = bf16(S[d][e]) (n-major for B frags)
  int bh = blockIdx.x;               // b*16 + h
  int b = bh >> 4, h = bh & 15;
  int tid = threadIdx.x;
  int lane = tid & 31, mr = lane & 15, half = lane >> 4, w = tid >> 5;
  int t0 = w * 16;
  for (int idx = tid; idx < D_ * D_; idx += 128) { S[idx] = 0.f; SbT[idx] = 0; }
  __syncthreads();
  for (int n = 0; n < NC_; ++n) {
    int cb = bh * NC_ + n;           // matches k3 blockIdx
    long base = ((long)b * L_ + n * C_) * DIM_ + h * D_;
    if (n > 0) {                     // inter = q~ @ S_start, add into out
      f32x8 acc[4] = {};
#pragma unroll
      for (int kk = 0; kk < D_; kk += 32) {
        bf16x16 a = loadA_rm(qtw + base, DIM_, t0, kk, mr, half);
#pragma unroll
        for (int j = 0; j < 4; ++j)
          acc[j] = wmma_bf16(a, loadB_cm(SbT, D_, kk, j * 16, mr, half), acc[j]);
      }
#pragma unroll
      for (int j = 0; j < 4; ++j)
#pragma unroll
        for (int r = 0; r < 8; ++r) {
          long o = base + (long)(t0 + r + half * 8) * DIM_ + j * 16 + mr;
          out[o] += acc[j][r];       // disjoint columns per (b,h): no races
        }
    }
    __syncthreads();
    for (int idx = tid; idx < D_ * D_; idx += 128) {  // S <- total(.)S + U
      int d = idx >> 6, e = idx & 63;
      float s = totw[cb * D_ + d] * S[idx] + Uw[(long)cb * (D_ * D_) + idx];
      S[idx] = s; SbT[e * D_ + d] = f2bf(s);
    }
    __syncthreads();
  }
}

// ---------------------------------------------------------------- launcher
extern "C" void kernel_launch(void* const* d_in, const int* in_sizes, int n_in,
                              void* d_out, int out_size, void* d_ws, size_t ws_size,
                              hipStream_t stream) {
  (void)in_sizes; (void)n_in; (void)out_size; (void)ws_size;
  const float* x   = (const float*)d_in[0];
  const float* Wq  = (const float*)d_in[1];
  const float* bq  = (const float*)d_in[2];
  const float* Wk  = (const float*)d_in[3];
  const float* bk  = (const float*)d_in[4];
  const float* Wv  = (const float*)d_in[5];
  const float* bv  = (const float*)d_in[6];
  const float* Wg1 = (const float*)d_in[7];
  const float* bg1 = (const float*)d_in[8];
  const float* Wg2 = (const float*)d_in[9];
  const float* bg2 = (const float*)d_in[10];
  float* out = (float*)d_out;

  // workspace layout (~78 MiB total)
  char* ws = (char*)d_ws;
  const size_t NX = (size_t)TOK * DIM_;                 // 4194304 elements
  u16* xb  = (u16*)ws;  ws += NX * 2;                   // x in bf16
  u16* wt  = (u16*)ws;  ws += (size_t)3 * DIM_ * DIM_ * 2;  // Wq/Wk/Wv transposed bf16
  u16* qb  = (u16*)ws;  ws += NX * 2;
  u16* kb  = (u16*)ws;  ws += NX * 2;
  u16* vb  = (u16*)ws;  ws += NX * 2;
  u16* qtw = (u16*)ws;  ws += NX * 2;                   // q (.) decay-cumprod
  float* alphaw = (float*)ws;  ws += NX * 4;
  float* Uw     = (float*)ws;  ws += (size_t)NCHUNK * D_ * D_ * 4;
  float* totw   = (float*)ws;  ws += (size_t)NCHUNK * D_ * 4;

  long ntot = (long)NX + 3l * DIM_ * DIM_;
  k0_convert<<<(int)((ntot + 255) / 256), 256, 0, stream>>>(x, Wq, Wk, Wv, xb, wt);
  k1_qkv<<<1536, 256, 0, stream>>>(xb, wt, bq, bk, bv, qb, kb, vb);
  k2_gate<<<TOK, 256, 0, stream>>>(x, Wg1, bg1, Wg2, bg2, alphaw);
  k3_chunk<<<NCHUNK, 128, 0, stream>>>(qb, kb, vb, alphaw, qtw, Uw, totw, out);
  k4_scan<<<B_ * H_, 128, 0, stream>>>(qtw, Uw, totw, out);
}
